// GATv2Backbone_4578435137605
// MI455X (gfx1250) — matrix operand order repaired and checked by reading, not compiled, not run on hardware
//
#include <hip/hip_runtime.h>
#include <hip/hip_bf16.h>
#include <math.h>

#define NN 50000
#define NE 800000
#define NG 50
#define HEADS 4
#define DHID 128
#define DOUT 128
#define NEG_SLOPE 0.2f
#define LN_EPS 1e-5f

typedef __attribute__((ext_vector_type(16))) __bf16 v16bf;
typedef __attribute__((ext_vector_type(8)))  float  v8f;

// ---------- helpers ----------
__device__ __forceinline__ unsigned fkey(float f) {
  unsigned u = __float_as_uint(f);
  return (u & 0x80000000u) ? ~u : (u | 0x80000000u);
}
__device__ __forceinline__ float fkey_dec(unsigned u) {
  return __uint_as_float((u & 0x80000000u) ? (u & 0x7FFFFFFFu) : ~u);
}

// ---------- weight prep: f32 [K][N] -> bf16 [N][K] ----------
__global__ void k_prep_w(const float* __restrict__ W, __bf16* __restrict__ Wt, int K, int N) {
  int i = blockIdx.x * blockDim.x + threadIdx.x;
  if (i >= K * N) return;
  int k = i / N, n = i % N;
  Wt[(size_t)n * K + k] = (__bf16)W[i];
}

// ---------- activation convert (+optional residual) ----------
__global__ void k_cvt(const float* __restrict__ h, const float* __restrict__ resid,
                      __bf16* __restrict__ out, int n) {
  int i = blockIdx.x * blockDim.x + threadIdx.x;
  if (i >= n) return;
  float v = h[i];
  if (resid) v += resid[i];
  out[i] = (__bf16)v;
}

// ---------- WMMA GEMM: C[M,N] = A[M,K=128](bf16) * Wt[N,K]^T(bf16) + bias[N] ----------
// One wave computes a 16x64 strip; fragments double-buffered across k-steps so the
// next step's loads are in flight during the current 4-WMMA burst.
union Frag { uint4 u[2]; v16bf v; };

__global__ __launch_bounds__(256)
void k_gemm(const __bf16* __restrict__ A, const __bf16* __restrict__ Wt,
            const float* __restrict__ bias, float* __restrict__ C,
            int M, int N) {
  const int lane   = threadIdx.x & 31;
  const int wave   = (blockIdx.x * blockDim.x + threadIdx.x) >> 5;
  const int tilesN = N >> 6;                       // strips of 64 columns
  const int tileM  = wave / tilesN;
  const int tileN  = wave - tileM * tilesN;
  if (tileM * 16 >= M) return;                     // wave-uniform: EXEC all-ones
  const int lo16 = lane & 15;
  const int hi   = lane >> 4;

  v8f acc[4];
#pragma unroll
  for (int j = 0; j < 4; ++j) {
    float bv = bias ? bias[tileN * 64 + j * 16 + lo16] : 0.0f;
#pragma unroll
    for (int r = 0; r < 8; ++r) acc[j][r] = bv;
  }

  // A frag (16x32 bf16): lane m=lo16; v0..v3 = k[8*hi..+7], v4..v7 = k[16+8*hi..+7]
  // B frag (32x16 bf16): lane n=lo16; v0..v3 = k[8*hi..+7], v4..v7 = k[16+8*hi..+7]
  const __bf16* pa  = A  + (size_t)(tileM * 16 + lo16) * 128;
  const __bf16* pb0 = Wt + (size_t)(tileN * 64 + lo16) * 128;

  Frag a[2], b[2][4];

  // prologue: load k-step 0 into buffer 0
  a[0].u[0] = *(const uint4*)(pa + 8 * hi);
  a[0].u[1] = *(const uint4*)(pa + 16 + 8 * hi);
#pragma unroll
  for (int j = 0; j < 4; ++j) {
    const __bf16* pb = pb0 + (size_t)(j * 16) * 128;
    b[0][j].u[0] = *(const uint4*)(pb + 8 * hi);
    b[0][j].u[1] = *(const uint4*)(pb + 16 + 8 * hi);
  }

#pragma unroll
  for (int s = 0; s < 4; ++s) {
    const int cur = s & 1, nxt = cur ^ 1;
    if (s < 3) {                                   // prefetch k-step s+1
      const int kb = (s + 1) * 32;
      a[nxt].u[0] = *(const uint4*)(pa + kb + 8 * hi);
      a[nxt].u[1] = *(const uint4*)(pa + kb + 16 + 8 * hi);
#pragma unroll
      for (int j = 0; j < 4; ++j) {
        const __bf16* pb = pb0 + (size_t)(j * 16) * 128;
        b[nxt][j].u[0] = *(const uint4*)(pb + kb + 8 * hi);
        b[nxt][j].u[1] = *(const uint4*)(pb + kb + 16 + 8 * hi);
      }
    }
#pragma unroll
    for (int j = 0; j < 4; ++j)
      acc[j] = __builtin_amdgcn_wmma_f32_16x16x32_bf16(false, a[cur].v, false, b[cur][j].v,
                                                       (short)0, acc[j], false, false);
  }

#pragma unroll
  for (int j = 0; j < 4; ++j) {
    float* pc = C + (size_t)(tileM * 16 + 8 * hi) * N + tileN * 64 + j * 16 + lo16;
#pragma unroll
    for (int r = 0; r < 8; ++r) pc[(size_t)r * N] = acc[j][r];
  }
}

// ---------- segment-softmax buffers init ----------
__global__ void k_init_seg(unsigned* __restrict__ gmax, float* __restrict__ gsum, int n) {
  int i = blockIdx.x * blockDim.x + threadIdx.x;
  if (i >= n) return;
  gmax[i] = 0x007FFFFFu;   // fkey(-inf)
  gsum[i] = 0.0f;
}

__device__ __forceinline__ float lrelu_dot4(float4 a, float4 b, float w, float4 we, float4 at) {
  float v0 = a.x + b.x + w * we.x; v0 = v0 > 0.f ? v0 : NEG_SLOPE * v0;
  float v1 = a.y + b.y + w * we.y; v1 = v1 > 0.f ? v1 : NEG_SLOPE * v1;
  float v2 = a.z + b.z + w * we.z; v2 = v2 > 0.f ? v2 : NEG_SLOPE * v2;
  float v3 = a.w + b.w + w * we.w; v3 = v3 > 0.f ? v3 : NEG_SLOPE * v3;
  return at.x * v0 + at.y * v1 + at.z * v2 + at.w * v3;
}

// ---------- per-(edge,head): logits + segment atomic max ----------
__global__ void k_edge_logits(const float* __restrict__ xl, const float* __restrict__ xr,
                              const int* __restrict__ src, const int* __restrict__ dst,
                              const float* __restrict__ ew, const float* __restrict__ att,
                              const float* __restrict__ We, float* __restrict__ logits,
                              unsigned* __restrict__ gmax, int C) {
  int t = blockIdx.x * blockDim.x + threadIdx.x;
  if (t >= NE * HEADS) return;
  int e = t / HEADS, h = t - e * HEADS;
  int F = HEADS * C;
  int s = src[e], d = dst[e];
  float w = ew[e];
  const float4* pl = (const float4*)(xl + (size_t)s * F + h * C);
  const float4* pr = (const float4*)(xr + (size_t)d * F + h * C);
  const float4* pe = (const float4*)(We + h * C);
  const float4* pa = (const float4*)(att + h * C);
  float acc = 0.f;
  for (int c = 0; c < (C >> 2); ++c)
    acc += lrelu_dot4(pl[c], pr[c], w, pe[c], pa[c]);
  logits[t] = acc;
  atomicMax(&gmax[(size_t)d * HEADS + h], fkey(acc));
}

// ---------- per-(edge,head): exp + segment atomic sum ----------
__global__ void k_edge_exp(const float* __restrict__ logits, const int* __restrict__ dst,
                           const unsigned* __restrict__ gmax, float* __restrict__ evals,
                           float* __restrict__ gsum) {
  int t = blockIdx.x * blockDim.x + threadIdx.x;
  if (t >= NE * HEADS) return;
  int e = t / HEADS, h = t - e * HEADS;
  int d = dst[e];
  float m = fkey_dec(gmax[(size_t)d * HEADS + h]);
  float ev = __expf(logits[t] - m);
  evals[t] = ev;
  atomicAdd(&gsum[(size_t)d * HEADS + h], ev);
}

// ---------- agg init with bias (or zero) ----------
__global__ void k_init_agg(float* __restrict__ agg, const float* __restrict__ bias,
                           int n, int F) {
  int i = blockIdx.x * blockDim.x + threadIdx.x;
  if (i >= n) return;
  agg[i] = bias ? bias[i % F] : 0.0f;
}

// ---------- per-(edge,head): alpha-weighted scatter-add ----------
__global__ void k_edge_agg(const float* __restrict__ xl, const int* __restrict__ src,
                           const int* __restrict__ dst, const float* __restrict__ evals,
                           const float* __restrict__ gsum, float* __restrict__ agg, int C) {
  int t = blockIdx.x * blockDim.x + threadIdx.x;
  if (t >= NE * HEADS) return;
  int e = t / HEADS, h = t - e * HEADS;
  int F = HEADS * C;
  int s = src[e], d = dst[e];
  float alpha = evals[t] / (gsum[(size_t)d * HEADS + h] + 1e-16f);
  const float4* px = (const float4*)(xl + (size_t)s * F + h * C);
  float* po = agg + (size_t)d * F + h * C;
  for (int c = 0; c < (C >> 2); ++c) {
    float4 v = px[c];
    atomicAdd(&po[4 * c + 0], alpha * v.x);
    atomicAdd(&po[4 * c + 1], alpha * v.y);
    atomicAdd(&po[4 * c + 2], alpha * v.z);
    atomicAdd(&po[4 * c + 3], alpha * v.w);
  }
}

// ---------- graph layernorm ----------
__global__ void k_zero_stats(float* __restrict__ gs, int n) {
  int i = blockIdx.x * blockDim.x + threadIdx.x;
  if (i < n) gs[i] = 0.0f;
}
__global__ void k_ln_node(const float* __restrict__ x, const int* __restrict__ batch,
                          float* __restrict__ gs) {
  int n = blockIdx.x * blockDim.x + threadIdx.x;
  if (n >= NN) return;
  const float4* p = (const float4*)(x + (size_t)n * DHID);
  float s1 = 0.f, s2 = 0.f;
  for (int f = 0; f < DHID / 4; ++f) {
    float4 v = p[f];
    s1 += v.x + v.y + v.z + v.w;
    s2 += v.x * v.x + v.y * v.y + v.z * v.z + v.w * v.w;
  }
  int g = batch[n];
  atomicAdd(&gs[g], s1);
  atomicAdd(&gs[NG + g], s2);
  atomicAdd(&gs[2 * NG + g], 1.0f);
}
__global__ void k_graph_stats(const float* __restrict__ gs, float* __restrict__ stats) {
  int g = threadIdx.x;
  if (g >= NG) return;
  float cnt = gs[2 * NG + g];
  float denom = fmaxf(cnt, 1.0f) * (float)DHID;
  float mean = gs[g] / denom;
  float var = gs[NG + g] / denom - mean * mean;
  stats[g] = mean;
  stats[NG + g] = rsqrtf(var + LN_EPS);
}
__global__ void k_ln_apply(const float* __restrict__ x, const int* __restrict__ batch,
                           const float* __restrict__ stats, const float* __restrict__ w,
                           const float* __restrict__ b, float* __restrict__ out) {
  int i = blockIdx.x * blockDim.x + threadIdx.x;
  if (i >= NN * DHID) return;
  int n = i / DHID, f = i - n * DHID;
  int g = batch[n];
  float v = (x[i] - stats[g]) * stats[NG + g] * w[f] + b[f];
  out[i] = v > 0.f ? v : (__expf(v) - 1.0f);   // ELU
}

// ---------- final head-mean + bias ----------
__global__ void k_head_mean(const float* __restrict__ agg, const float* __restrict__ bias,
                            float* __restrict__ out) {
  int i = blockIdx.x * blockDim.x + threadIdx.x;
  if (i >= NN * DOUT) return;
  int n = i / DOUT, c = i - n * DOUT;
  const float* p = agg + (size_t)n * (HEADS * DOUT) + c;
  out[i] = 0.25f * (p[0] + p[DOUT] + p[2 * DOUT] + p[3 * DOUT]) + bias[c];
}

// ===================== launch =====================
static inline int cdiv(long long a, long long b) { return (int)((a + b - 1) / b); }

extern "C" void kernel_launch(void* const* d_in, const int* in_sizes, int n_in,
                              void* d_out, int out_size, void* d_ws, size_t ws_size,
                              hipStream_t stream) {
  const float* x      = (const float*)d_in[0];
  const int*   ei     = (const int*)d_in[1];
  const float* ew     = (const float*)d_in[2];
  const int*   batch  = (const int*)d_in[3];
  const float* resid  = (const float*)d_in[4];
  const float* Wl     = (const float*)d_in[5];
  const float* Wr     = (const float*)d_in[6];
  const float* bl     = (const float*)d_in[7];
  const float* br     = (const float*)d_in[8];
  const float* att    = (const float*)d_in[9];
  const float* We     = (const float*)d_in[10];
  const float* ln_w   = (const float*)d_in[11];
  const float* ln_b   = (const float*)d_in[12];
  const float* bias   = (const float*)d_in[13];
  const float* Wl_f   = (const float*)d_in[14];
  const float* Wr_f   = (const float*)d_in[15];
  const float* bl_f   = (const float*)d_in[16];
  const float* br_f   = (const float*)d_in[17];
  const float* att_f  = (const float*)d_in[18];
  const float* We_f   = (const float*)d_in[19];
  const float* bias_f = (const float*)d_in[20];
  const int* src = ei;
  const int* dst = ei + NE;

  // workspace carve-up
  char* ws = (char*)d_ws;
  size_t off = 0;
  auto carve = [&](size_t bytes) { char* p = ws + off; off = (off + bytes + 255) & ~(size_t)255; return p; };
  float*    hcur  = (float*)   carve((size_t)NN * DHID * 4);
  __bf16*   hbf   = (__bf16*)  carve((size_t)NN * DHID * 2);
  float*    xl    = (float*)   carve((size_t)NN * 512 * 4);
  float*    xr    = (float*)   carve((size_t)NN * 512 * 4);
  float*    agg   = (float*)   carve((size_t)NN * 512 * 4);
  float*    logit = (float*)   carve((size_t)NE * HEADS * 4);
  float*    evals = (float*)   carve((size_t)NE * HEADS * 4);
  unsigned* gmax  = (unsigned*)carve((size_t)NN * HEADS * 4);
  float*    gsum  = (float*)   carve((size_t)NN * HEADS * 4);
  float*    gs    = (float*)   carve(3 * NG * 4);
  float*    stats = (float*)   carve(2 * NG * 4);
  __bf16*   Wtl   = (__bf16*)  carve((size_t)512 * 128 * 2);
  __bf16*   Wtr   = (__bf16*)  carve((size_t)512 * 128 * 2);

  const int B = 256;
  const int EH = NE * HEADS;

  // ---- layers 0..2 (width 128, concat) ----
  for (int i = 0; i < 3; ++i) {
    const float* hin = (i == 0) ? x : hcur;
    k_prep_w<<<cdiv(128 * 128, B), B, 0, stream>>>(Wl + (size_t)i * 128 * 128, Wtl, 128, 128);
    k_prep_w<<<cdiv(128 * 128, B), B, 0, stream>>>(Wr + (size_t)i * 128 * 128, Wtr, 128, 128);
    k_cvt<<<cdiv((long long)NN * 128, B), B, 0, stream>>>(hin, (i == 2) ? resid : nullptr, hbf, NN * 128);

    int waves = (NN / 16) * (128 / 64);            // 6250 strip-waves
    k_gemm<<<cdiv(waves, 8), B, 0, stream>>>(hbf, Wtl, bl + (size_t)i * 128, xl, NN, 128);
    k_gemm<<<cdiv(waves, 8), B, 0, stream>>>(hbf, Wtr, br + (size_t)i * 128, xr, NN, 128);

    k_init_seg<<<cdiv(NN * HEADS, B), B, 0, stream>>>(gmax, gsum, NN * HEADS);
    k_edge_logits<<<cdiv(EH, B), B, 0, stream>>>(xl, xr, src, dst, ew,
                                                 att + (size_t)i * HEADS * 32,
                                                 We + (size_t)i * 128, logit, gmax, 32);
    k_edge_exp<<<cdiv(EH, B), B, 0, stream>>>(logit, dst, gmax, evals, gsum);
    k_init_agg<<<cdiv((long long)NN * 128, B), B, 0, stream>>>(agg, bias + (size_t)i * 128, NN * 128, 128);
    k_edge_agg<<<cdiv(EH, B), B, 0, stream>>>(xl, src, dst, evals, gsum, agg, 32);

    k_zero_stats<<<1, 3 * NG, 0, stream>>>(gs, 3 * NG);
    k_ln_node<<<cdiv(NN, B), B, 0, stream>>>(agg, batch, gs);
    k_graph_stats<<<1, 64, 0, stream>>>(gs, stats);
    k_ln_apply<<<cdiv((long long)NN * 128, B), B, 0, stream>>>(agg, batch, stats,
                                                               ln_w + (size_t)i * 128,
                                                               ln_b + (size_t)i * 128, hcur);
  }

  // ---- final layer (width 512, mean over heads) ----
  k_prep_w<<<cdiv(128 * 512, B), B, 0, stream>>>(Wl_f, Wtl, 128, 512);
  k_prep_w<<<cdiv(128 * 512, B), B, 0, stream>>>(Wr_f, Wtr, 128, 512);
  k_cvt<<<cdiv((long long)NN * 128, B), B, 0, stream>>>(hcur, nullptr, hbf, NN * 128);

  int waves_f = (NN / 16) * (512 / 64);            // 25000 strip-waves
  k_gemm<<<cdiv(waves_f, 8), B, 0, stream>>>(hbf, Wtl, bl_f, xl, NN, 512);
  k_gemm<<<cdiv(waves_f, 8), B, 0, stream>>>(hbf, Wtr, br_f, xr, NN, 512);

  k_init_seg<<<cdiv(NN * HEADS, B), B, 0, stream>>>(gmax, gsum, NN * HEADS);
  k_edge_logits<<<cdiv(EH, B), B, 0, stream>>>(xl, xr, src, dst, ew, att_f, We_f, logit, gmax, 128);
  k_edge_exp<<<cdiv(EH, B), B, 0, stream>>>(logit, dst, gmax, evals, gsum);
  k_init_agg<<<cdiv((long long)NN * 512, B), B, 0, stream>>>(agg, nullptr, NN * 512, 512);
  k_edge_agg<<<cdiv(EH, B), B, 0, stream>>>(xl, src, dst, evals, gsum, agg, 128);
  k_head_mean<<<cdiv((long long)NN * DOUT, B), B, 0, stream>>>(agg, bias_f, (float*)d_out);

  (void)in_sizes; (void)n_in; (void)out_size; (void)ws_size;
}
